// SparseMoEConfidNet_72834055405854
// MI455X (gfx1250) — matrix operand order repaired and checked by reading, not compiled, not run on hardware
//
#include <hip/hip_runtime.h>
#include <hip/hip_bf16.h>

typedef __attribute__((ext_vector_type(16))) _Float16 v16h;
typedef __attribute__((ext_vector_type(8)))  _Float16 v8h;
typedef __attribute__((ext_vector_type(4)))  _Float16 v4h;
typedef __attribute__((ext_vector_type(8)))  float    v8f;

#define B_   65536
#define H_   768
#define V_   512
#define D_   1280
#define E_   8
#define N1_  256
#define N2_  128
#define MT_  32          // rows per block (two 16-row WMMA M-tiles)

// LDS row strides (halves), padded so row stride mod 64 banks = 4 dwords
#define XS_  (D_ + 8)    // 1288
#define H1S_ (N1_ + 8)   // 264
#define H2S_ (N2_ + 8)   // 136

// -------------------- weight convert/transpose: fp32 [e][K][N] -> f16 [e][N][K]
__global__ void transpose_to_f16(const float* __restrict__ src, _Float16* __restrict__ dst,
                                 int Eb, int K, int N) {
    size_t i = (size_t)blockIdx.x * blockDim.x + threadIdx.x;
    size_t total = (size_t)Eb * K * N;
    if (i >= total) return;
    int kn = K * N;
    int e = (int)(i / kn);
    int r = (int)(i % kn);
    int k = r / N;
    int n = r % N;
    dst[((size_t)e * N + n) * K + k] = (_Float16)src[i];
}

// -------------------- WMMA fragment loaders
// A fragment: 16x32 f16 tile from LDS, row-major, stride `lda` halves.
// ISA layout: lane<16 -> M=lane, K = {k0..k0+7, k0+16..k0+23}
//             lane>=16 -> M=lane-16, K = {k0+8..k0+15, k0+24..k0+31}
__device__ __forceinline__ v16h load_a_frag(const _Float16* a, int lda, int k0, int lane) {
    int row  = lane & 15;
    int koff = (lane < 16) ? 0 : 8;
    const _Float16* p = a + row * lda + k0 + koff;
    union { v16h v; v8h h[2]; } u;
    u.h[0] = *(const v8h*)(p);        // K = k0+koff .. +7
    u.h[1] = *(const v8h*)(p + 16);   // K = k0+koff+16 .. +23
    return u.v;
}

// B fragment: 32x16 f16 tile, weights pre-transposed to [N][K] (K contiguous).
// lane<16 -> N=n0+lane, K={k0..k0+7, k0+16..k0+23} ; lane>=16 -> N=n0+lane-16, K={k0+8..k0+15, k0+24..k0+31}
__device__ __forceinline__ v16h load_b_frag(const _Float16* __restrict__ w, int ldb,
                                            int n0, int k0, int lane) {
    int n  = n0 + (lane & 15);
    int kb = k0 + ((lane < 16) ? 0 : 8);
    const _Float16* p = w + (size_t)n * ldb + kb;
    union { v16h v; v8h h[2]; } u;
    u.h[0] = *(const v8h*)(p);
    u.h[1] = *(const v8h*)(p + 16);
    return u.v;
}

#define WMMA_F16(a, b, c) __builtin_amdgcn_wmma_f32_16x16x32_f16(false, (a), false, (b), \
                                                                 (short)0, (c), false, false)

// 2M x 2N register tile: rows [0,32) x cols [n0,n0+32) of relu(A(32xK) @ W^T + bias).
// Each B fragment feeds two M-tiles (halves L2 weight traffic); each A fragment feeds
// two N-tiles (halves LDS traffic). 4 independent WMMAs per k-step.
template<int K>
__device__ __forceinline__ void gemm22_relu(const _Float16* a_lds, int lda,
                                            const _Float16* __restrict__ w_t,  // [N][K] f16
                                            const float* __restrict__ bias,    // [N] f32
                                            _Float16* c_lds, int ldc,
                                            int n0, int lane) {
    v8f a00 = {}, a01 = {}, a10 = {}, a11 = {};
#pragma unroll 2
    for (int k0 = 0; k0 < K; k0 += 32) {
        v16h afA = load_a_frag(a_lds,            lda, k0, lane);   // rows  0..15
        v16h afB = load_a_frag(a_lds + 16 * lda, lda, k0, lane);   // rows 16..31
        v16h b0  = load_b_frag(w_t, K, n0,      k0, lane);
        v16h b1  = load_b_frag(w_t, K, n0 + 16, k0, lane);
        __builtin_prefetch(w_t + (size_t)(n0 + (lane & 15)) * K + k0 + 128, 0, 1);
        a00 = WMMA_F16(afA, b0, a00);
        a10 = WMMA_F16(afB, b0, a10);
        a01 = WMMA_F16(afA, b1, a01);
        a11 = WMMA_F16(afB, b1, a11);
    }
    // C/D layout: VGPR r -> lanes 0-15: (M=r, N=lane) ; lanes 16-31: (M=8+r, N=lane-16)
    int nA    = n0 + (lane & 15);
    int mbase = (lane < 16) ? 0 : 8;
    float bA  = bias[nA];
    float bB  = bias[nA + 16];
#pragma unroll
    for (int r = 0; r < 8; ++r) {
        float v00 = a00[r] + bA; v00 = v00 > 0.0f ? v00 : 0.0f;
        float v01 = a01[r] + bB; v01 = v01 > 0.0f ? v01 : 0.0f;
        float v10 = a10[r] + bA; v10 = v10 > 0.0f ? v10 : 0.0f;
        float v11 = a11[r] + bB; v11 = v11 > 0.0f ? v11 : 0.0f;
        c_lds[(mbase + r) * ldc + nA]             = (_Float16)v00;
        c_lds[(mbase + r) * ldc + nA + 16]        = (_Float16)v01;
        c_lds[(mbase + r + 16) * ldc + nA]        = (_Float16)v10;
        c_lds[(mbase + r + 16) * ldc + nA + 16]   = (_Float16)v11;
    }
}

// Router layer 2 ((32x256)@(256x8)) + top-2 + softmax over 32 rows. Whole wave active.
__device__ __forceinline__ void router_tail32(const _Float16* h, const float* __restrict__ rw2,
                                              const float* __restrict__ rb2,
                                              float* lg, float* wts, int* idx, int lane) {
#pragma unroll
    for (int i = 0; i < 8; ++i) {
        int p = lane + 32 * i;          // 0..255 = 32 rows x 8 experts
        int m = p >> 3, e = p & 7;
        float s = rb2[e];
        for (int k = 0; k < N1_; ++k)
            s += (float)h[m * H1S_ + k] * rw2[k * E_ + e];
        lg[m * E_ + e] = s;
    }
    int m = lane;                        // one row per lane
    float b0 = -1e30f, b1v = -1e30f; int i0 = 0, i1 = 0;
    for (int e = 0; e < E_; ++e) {
        float v = lg[m * E_ + e];
        if (v > b0)       { b1v = b0; i1 = i0; b0 = v; i0 = e; }
        else if (v > b1v) { b1v = v; i1 = e; }
    }
    float e1  = __expf(b1v - b0);
    float inv = 1.0f / (1.0f + e1);
    wts[m * 2]     = inv;
    wts[m * 2 + 1] = e1 * inv;
    idx[m * 2]     = i0;
    idx[m * 2 + 1] = i1;
}

// -------------------- fused MoE-ConfidNet kernel: one block = 32 rows, 8 waves
__global__ __launch_bounds__(256)
void moe_confidnet_kernel(const float* __restrict__ text, const float* __restrict__ video,
                          const _Float16* __restrict__ w1t,   // [E][256][1280] f16
                          const _Float16* __restrict__ w2t,   // [E][128][256]  f16
                          const float* __restrict__ exp_b1,   // [E][256]
                          const float* __restrict__ exp_b2,   // [E][128]
                          const float* __restrict__ exp_w3,   // [E][128]
                          const float* __restrict__ exp_b3,   // [E]
                          const _Float16* __restrict__ t1t,   // [256][768] f16
                          const float* __restrict__ t_b1, const float* __restrict__ t_w2,
                          const float* __restrict__ t_b2,
                          const _Float16* __restrict__ v1t,   // [256][512] f16
                          const float* __restrict__ v_b1, const float* __restrict__ v_w2,
                          const float* __restrict__ v_b2,
                          float* __restrict__ out) {
    extern __shared__ char smem[];
    _Float16* Xs     = (_Float16*)smem;                     // [32][XS_]
    _Float16* h1base = Xs + MT_ * XS_;                      // 8 x [32][H1S_]
    _Float16* h2base = h1base + 8 * MT_ * H1S_;             // 8 x [32][H2S_]
    float*    eo     = (float*)(h2base + 8 * MT_ * H2S_);   // [32][8]
    float*    tlog   = eo + MT_ * E_;                       // [32][8]
    float*    vlog   = tlog + MT_ * E_;                     // [32][8]
    float*    tw     = vlog + MT_ * E_;                     // [32][2]
    float*    vw     = tw + MT_ * 2;                        // [32][2]
    int*      tidx   = (int*)(vw + MT_ * 2);                // [32][2]
    int*      vidx   = tidx + MT_ * 2;                      // [32][2]

    const int tid  = threadIdx.x;
    const int wid  = tid >> 5;          // 0..7 : one expert per wave
    const int lane = tid & 31;
    const int row0 = blockIdx.x * MT_;

    _Float16* myh1 = h1base + wid * MT_ * H1S_;
    _Float16* myh2 = h2base + wid * MT_ * H2S_;

    // ---- stage combined = [text | video] as f16 into LDS (vectorized x4)
    {
        const int HQ = H_ / 4;
        const float4* t4 = (const float4*)text;
        for (int i = tid; i < MT_ * HQ; i += 256) {
            int r = i / HQ, q = i % HQ;
            float4 v = t4[(size_t)(row0 + r) * HQ + q];
            v4h h; h.x = (_Float16)v.x; h.y = (_Float16)v.y;
                   h.z = (_Float16)v.z; h.w = (_Float16)v.w;
            *(v4h*)(&Xs[r * XS_ + 4 * q]) = h;
        }
        const int VQ = V_ / 4;
        const float4* s4 = (const float4*)video;
        for (int i = tid; i < MT_ * VQ; i += 256) {
            int r = i / VQ, q = i % VQ;
            float4 v = s4[(size_t)(row0 + r) * VQ + q];
            v4h h; h.x = (_Float16)v.x; h.y = (_Float16)v.y;
                   h.z = (_Float16)v.z; h.w = (_Float16)v.w;
            *(v4h*)(&Xs[r * XS_ + H_ + 4 * q]) = h;
        }
    }
    __syncthreads();

    // ---- router layer 1, distributed: wave w computes N-columns [32w,32w+32)
    //      of BOTH routers (balanced across all 8 waves).
    //      Text h1 -> buffer 0, video h1 -> buffer 1 (recycled below by waves 0/1).
    gemm22_relu<H_>(Xs,      XS_, t1t, t_b1, h1base,              H1S_, wid * 32, lane);
    gemm22_relu<V_>(Xs + H_, XS_, v1t, v_b1, h1base + MT_ * H1S_, H1S_, wid * 32, lane);
    __syncthreads();

    // ---- router tails (tiny): wave 0 text, wave 1 video
    if (wid == 0)      router_tail32(h1base,              t_w2, t_b2, tlog, tw, tidx, lane);
    else if (wid == 1) router_tail32(h1base + MT_ * H1S_, v_w2, v_b2, vlog, vw, vidx, lane);

    // ---- expert wid (all-dense, matches reference einsum); waves 0/1 reuse the
    //      router h1 buffers as their private expert h1 (only owner touches them now).
    {
        const int e = wid;
        const _Float16* w1e = w1t + (size_t)e * N1_ * D_;
        const _Float16* w2e = w2t + (size_t)e * N2_ * N1_;

#pragma unroll 1
        for (int np = 0; np < N1_ / 32; ++np)       // 32x256 = relu(X @ W1 + b1)
            gemm22_relu<D_>(Xs, XS_, w1e, exp_b1 + e * N1_, myh1, H1S_, np * 32, lane);
#pragma unroll 1
        for (int np = 0; np < N2_ / 32; ++np)       // 32x128 = relu(h1 @ W2 + b2)
            gemm22_relu<N1_>(myh1, H1S_, w2e, exp_b2 + e * N2_, myh2, H2S_, np * 32, lane);

        // sigmoid(h2 . w3 + b3): one row per lane
        const float* w3 = exp_w3 + e * N2_;
        float s = exp_b3[e];
        for (int f = 0; f < N2_; ++f)
            s += (float)myh2[lane * H2S_ + f] * w3[f];
        eo[lane * E_ + e] = 1.0f / (1.0f + __expf(-s));
    }
    __syncthreads();

    // ---- aggregate top-2: wave 0 -> text rows, wave 1 -> video rows (1 row/lane)
    if (wid == 0) {
        int m = lane;
        float c = tw[m * 2]     * eo[m * E_ + tidx[m * 2]] +
                  tw[m * 2 + 1] * eo[m * E_ + tidx[m * 2 + 1]];
        out[row0 + m] = c;
    } else if (wid == 1) {
        int m = lane;
        float c = vw[m * 2]     * eo[m * E_ + vidx[m * 2]] +
                  vw[m * 2 + 1] * eo[m * E_ + vidx[m * 2 + 1]];
        out[(size_t)B_ + row0 + m] = c;
    }
}

// -------------------- host launcher
extern "C" void kernel_launch(void* const* d_in, const int* in_sizes, int n_in,
                              void* d_out, int out_size, void* d_ws, size_t ws_size,
                              hipStream_t stream) {
    const float* text   = (const float*)d_in[0];
    const float* video  = (const float*)d_in[1];
    const float* exp_w1 = (const float*)d_in[2];
    const float* exp_b1 = (const float*)d_in[3];
    const float* exp_w2 = (const float*)d_in[4];
    const float* exp_b2 = (const float*)d_in[5];
    const float* exp_w3 = (const float*)d_in[6];
    const float* exp_b3 = (const float*)d_in[7];
    const float* t_w1   = (const float*)d_in[8];
    const float* t_b1   = (const float*)d_in[9];
    const float* t_w2   = (const float*)d_in[10];
    const float* t_b2   = (const float*)d_in[11];
    const float* v_w1   = (const float*)d_in[12];
    const float* v_b1   = (const float*)d_in[13];
    const float* v_w2   = (const float*)d_in[14];
    const float* v_b2   = (const float*)d_in[15];

    _Float16* ws  = (_Float16*)d_ws;
    _Float16* w1t = ws;                                   // [8][256][1280]
    _Float16* w2t = w1t + (size_t)E_ * N1_ * D_;          // [8][128][256]
    _Float16* t1t = w2t + (size_t)E_ * N2_ * N1_;         // [256][768]
    _Float16* v1t = t1t + (size_t)N1_ * H_;               // [256][512]

    auto xpose = [&](const float* s, _Float16* d, int Eb, int K, int N) {
        size_t total = (size_t)Eb * K * N;
        int thr = 256;
        unsigned blk = (unsigned)((total + thr - 1) / thr);
        hipLaunchKernelGGL(transpose_to_f16, dim3(blk), dim3(thr), 0, stream, s, d, Eb, K, N);
    };
    xpose(exp_w1, w1t, E_, D_,  N1_);
    xpose(exp_w2, w2t, E_, N1_, N2_);
    xpose(t_w1,   t1t, 1,  H_,  N1_);
    xpose(v_w1,   v1t, 1,  V_,  N1_);

    size_t smem = (size_t)(MT_ * XS_ + 8 * MT_ * H1S_ + 8 * MT_ * H2S_) * sizeof(_Float16)
                + (size_t)(3 * MT_ * E_ + 4 * MT_) * sizeof(float)
                + (size_t)(4 * MT_) * sizeof(int);

    hipLaunchKernelGGL(moe_confidnet_kernel, dim3(B_ / MT_), dim3(256), smem, stream,
                       text, video, w1t, w2t, exp_b1, exp_b2, exp_w3, exp_b3,
                       t1t, t_b1, t_w2, t_b2, v1t, v_b1, v_w2, v_b2,
                       (float*)d_out);
}
